// CausalSelfAttention_2001454760112
// MI455X (gfx1250) — matrix operand order repaired and checked by reading, not compiled, not run on hardware
//
#include <hip/hip_runtime.h>

// ---------------------------------------------------------------------------
// CDNA5 (gfx1250) causal self-attention: bf16 WMMA + async-to-LDS staging.
// Wave32; WMMA 16x16x32 bf16 per CDNA5 ISA layouts.
// ---------------------------------------------------------------------------

typedef __bf16 bf16;
typedef __attribute__((ext_vector_type(16))) __bf16 v16bf;
typedef __attribute__((ext_vector_type(8)))  __bf16 v8bf;
typedef __attribute__((ext_vector_type(8)))  float  v8f;
typedef __attribute__((ext_vector_type(4)))  float  v4f;
typedef __attribute__((address_space(3)))    bf16   lds_bf16;

constexpr int Bc = 2, Tc = 2048, Dc = 1024, Hc = 16, HDc = 64;
constexpr int Mrows = Bc * Tc;     // 4096 rows for the projection GEMMs
constexpr int MT    = Mrows / 16;  // 256 row tiles

// A-fragment (16x32 bf16): per lane, two contiguous 8-element K-chunks.
union AFrag { v16bf v; v8bf h[2]; };

__device__ __forceinline__ v8f wmma_bf16(v16bf a, v16bf b, v8f c) {
  // (neg_a, A, neg_b, B, c_mod, C, reuse_a, reuse_b)
  return __builtin_amdgcn_wmma_f32_16x16x32_bf16(false, a, false, b, (short)0, c,
                                                 false, false);
}

// Async DMA: 16B global -> LDS per lane; completion tracked via ASYNCcnt.
__device__ __forceinline__ void async_copy_b128(const bf16* gsrc, bf16* lds_dst) {
  unsigned ldsoff = (unsigned)(unsigned long long)(lds_bf16*)lds_dst;
  unsigned long long gaddr = (unsigned long long)gsrc;
  asm volatile("global_load_async_to_lds_b128 %0, %1, off"
               :: "v"(ldsoff), "v"(gaddr) : "memory");
}

// ---------------------------------------------------------------------------
// Weight prep: W (f32, row-major DxD) -> W^T (bf16, row-major DxD).
// ---------------------------------------------------------------------------
__global__ __launch_bounds__(256) void wcvt_kernel(const float* __restrict__ W,
                                                   bf16* __restrict__ Wt) {
  const int i = blockIdx.x * 256 + threadIdx.x;   // over Dc*Dc
  const int k = i >> 10;
  const int n = i & (Dc - 1);
  Wt[(size_t)n * Dc + k] = (bf16)W[i];
}

// ---------------------------------------------------------------------------
// GEMM: one wave computes a 16x64 output tile; all 8 waves of a block share
// the same 64-wide column tile, so the B chunk (64x32 bf16 = 4KB) is staged
// cooperatively into LDS with GLOBAL_LOAD_ASYNC_TO_LDS_B128, double-buffered.
// AKIND: 0 = f32 A (convert on the fly), 1 = bf16 A.
// STORE: 0 = bf16 [bh][t][hd] (Q/K), 1 = bf16 [bh][hd][t] (V transposed),
//        2 = f32 row-major (final output).
// ---------------------------------------------------------------------------
template<int AKIND, int STORE>
__global__ __launch_bounds__(256) void gemm16x64_kernel(
    const void* __restrict__ Aptr, const bf16* __restrict__ Bt,
    const float* __restrict__ bias, void* __restrict__ Out) {
  __shared__ bf16 Bs[2][64 * 32];             // double-buffered B tile
  const int tid  = threadIdx.x;
  const int lane = tid & 31;
  const int wid  = tid >> 5;
  const int wg   = blockIdx.x * 8 + wid;      // 4096 wave tiles total
  const int mt   = wg & (MT - 1);             // row tile: varies per wave
  const int ct   = blockIdx.x >> 5;           // col tile: uniform per block
  const int row0 = mt * 16, col0 = ct * 64;
  const int ln   = lane & 15;                 // A row / B,C column within tile
  const int half = lane >> 4;                 // K-group select per ISA layout
  const int arow = row0 + ln;

  // Cooperative B staging: thread -> (column c, 16B K-chunk kk).
  const int c  = tid >> 2;                    // 0..63
  const int kk = (tid & 3) * 8;               // 0,8,16,24 (bf16 elements)
  const bf16* gsrc = Bt + (size_t)(col0 + c) * Dc + kk;

  v8f acc[4] = {};
  async_copy_b128(gsrc, &Bs[0][c * 32 + kk]); // prime chunk 0

  for (int k0 = 0; k0 < Dc; k0 += 32) {
    const int cur = (k0 >> 5) & 1;
    if (k0 + 32 < Dc) {                       // overlap next DMA with compute
      async_copy_b128(gsrc + k0 + 32, &Bs[cur ^ 1][c * 32 + kk]);
      asm volatile("s_wait_asynccnt 1" ::: "memory");
    } else {
      asm volatile("s_wait_asynccnt 0" ::: "memory");
    }
    __syncthreads();                          // B chunk visible block-wide

    AFrag a;
    if constexpr (AKIND == 0) {
      const float* ap = (const float*)Aptr + (size_t)arow * Dc + k0 + half * 16;
      v4f f0 = *(const v4f*)(ap);
      v4f f1 = *(const v4f*)(ap + 4);
      v4f g0 = *(const v4f*)(ap + 8);
      v4f g1 = *(const v4f*)(ap + 12);
#pragma unroll
      for (int i = 0; i < 4; ++i) {
        a.v[i]      = (bf16)f0[i];
        a.v[4 + i]  = (bf16)f1[i];
        a.v[8 + i]  = (bf16)g0[i];
        a.v[12 + i] = (bf16)g1[i];
      }
      if (k0 + 32 < Dc) __builtin_prefetch((const void*)(ap + 32), 0, 3);
    } else {
      const bf16* ap = (const bf16*)Aptr + (size_t)arow * Dc + k0 + half * 16;
      a.h[0] = *(const v8bf*)(ap);
      a.h[1] = *(const v8bf*)(ap + 8);
      if (k0 + 32 < Dc) __builtin_prefetch((const void*)(ap + 32), 0, 3);
    }

    const bf16* bp = &Bs[cur][half * 16];     // K-group base in LDS
#pragma unroll
    for (int nt = 0; nt < 4; ++nt) {
      v16bf bfrag = *(const v16bf*)(bp + (nt * 16 + ln) * 32);
      acc[nt] = wmma_bf16(a.v, bfrag, acc[nt]);
    }
    __syncthreads();                          // protect buffer reuse
  }

  // Epilogue: C layout element (m, n): m = r + half*8, n = ln.
#pragma unroll
  for (int nt = 0; nt < 4; ++nt) {
    const int gcol = col0 + nt * 16 + ln;
    const float bv = bias[gcol];
#pragma unroll
    for (int r = 0; r < 8; ++r) {
      const int grow = row0 + r + half * 8;
      const float v = acc[nt][r] + bv;
      if constexpr (STORE == 2) {
        ((float*)Out)[(size_t)grow * Dc + gcol] = v;
      } else {
        const int bb = grow / Tc, t = grow % Tc;
        const int h = gcol >> 6, hd = gcol & 63;
        if constexpr (STORE == 0)
          ((bf16*)Out)[(((size_t)(bb * Hc + h) * Tc + t) << 6) + hd] = (bf16)v;
        else
          ((bf16*)Out)[((size_t)(bb * Hc + h) * HDc + hd) * Tc + t] = (bf16)v;
      }
    }
  }
}

// ---------------------------------------------------------------------------
// RoPE on Q and K (bf16, layout [bh][t][64]); freqs = angle[t][j], j<32.
// ---------------------------------------------------------------------------
__global__ __launch_bounds__(256) void rope_kernel(bf16* __restrict__ Q,
                                                   bf16* __restrict__ K,
                                                   const float* __restrict__ freqs) {
  const int i = blockIdx.x * 256 + threadIdx.x;   // over B*H*T*32
  const int j = i & 31;
  const int t = (i >> 5) & (Tc - 1);
  const size_t off = ((size_t)(i >> 5)) * HDc + 2 * j;
  const float ang = freqs[t * 32 + j];
  const float c = __cosf(ang), s = __sinf(ang);
  float q0 = (float)Q[off], q1 = (float)Q[off + 1];
  Q[off]     = (bf16)(q0 * c - q1 * s);
  Q[off + 1] = (bf16)(q0 * s + q1 * c);
  float k0 = (float)K[off], k1 = (float)K[off + 1];
  K[off]     = (bf16)(k0 * c - k1 * s);
  K[off + 1] = (bf16)(k0 * s + k1 * c);
}

// ---------------------------------------------------------------------------
// Flash attention. 1 wave = 16 query rows; stream 32 keys per step.
// Q,K: [bh][t][64] bf16.  Vt: [bh][64][t] bf16.  AO: bf16 [b*T][D].
// ---------------------------------------------------------------------------
__global__ __launch_bounds__(128) void attn_kernel(const bf16* __restrict__ Q,
                                                   const bf16* __restrict__ K,
                                                   const bf16* __restrict__ Vt,
                                                   bf16* __restrict__ AO) {
  __shared__ bf16 Pbuf[4][16 * 32];                // per-wave P tile
  const int lane = threadIdx.x & 31;
  const int wid  = threadIdx.x >> 5;
  const int qb   = blockIdx.x & 31;                // T/64 = 32 query blocks
  const int bh   = blockIdx.x >> 5;                // B*H = 32
  const int qt0  = qb * 64 + wid * 16;             // this wave's query tile
  const int ln   = lane & 15, half = lane >> 4;

  // Q A-fragments (K-dim = head dim 64 -> two 16x32 fragments).
  const size_t qoff = ((size_t)bh * Tc + (qt0 + ln)) * HDc;
  AFrag qa0, qa1;
  qa0.h[0] = *(const v8bf*)(Q + qoff + half * 16);
  qa0.h[1] = *(const v8bf*)(Q + qoff + half * 16 + 8);
  qa1.h[0] = *(const v8bf*)(Q + qoff + 32 + half * 16);
  qa1.h[1] = *(const v8bf*)(Q + qoff + 32 + half * 16 + 8);

  v8f o[4] = {};
  float mrow[8], lrow[8];
#pragma unroll
  for (int r = 0; r < 8; ++r) { mrow[r] = -3.0e38f; lrow[r] = 0.0f; }
  bf16* P = Pbuf[wid];
  const float scale = 0.125f;                      // 1/sqrt(64)

  for (int j0 = 0; j0 <= qt0; j0 += 32) {          // causal: keys <= qt0+15
    // Scores: two 16x16 tiles over this 32-key chunk.
    v8f s0 = {}, s1 = {};
    {
      const bf16* kp0 = K + ((size_t)bh * Tc + (j0 + ln)) * HDc + half * 16;
      s0 = wmma_bf16(qa0.v, *(const v16bf*)kp0, s0);
      s0 = wmma_bf16(qa1.v, *(const v16bf*)(kp0 + 32), s0);
      const bf16* kp1 = K + ((size_t)bh * Tc + (j0 + 16 + ln)) * HDc + half * 16;
      s1 = wmma_bf16(qa0.v, *(const v16bf*)kp1, s1);
      s1 = wmma_bf16(qa1.v, *(const v16bf*)(kp1 + 32), s1);
    }
    // Mask + online softmax. Row m = r + half*8 lives in one 16-lane half.
#pragma unroll
    for (int r = 0; r < 8; ++r) {
      const int mr = qt0 + r + half * 8;
      float v0 = s0[r] * scale;
      float v1 = s1[r] * scale;
      if (j0 + ln > mr)      v0 = -1.0e30f;
      if (j0 + 16 + ln > mr) v1 = -1.0e30f;
      float mx = fmaxf(v0, v1);
#pragma unroll
      for (int d = 1; d < 16; d <<= 1) mx = fmaxf(mx, __shfl_xor(mx, d, 32));
      const float newm = fmaxf(mrow[r], mx);
      const float corr = __expf(mrow[r] - newm);
      const float p0 = __expf(v0 - newm);
      const float p1 = __expf(v1 - newm);
      float rs = p0 + p1;
#pragma unroll
      for (int d = 1; d < 16; d <<= 1) rs += __shfl_xor(rs, d, 32);
      lrow[r] = lrow[r] * corr + rs;
      mrow[r] = newm;
#pragma unroll
      for (int nt = 0; nt < 4; ++nt) o[nt][r] *= corr;
      const int mm = r + half * 8;
      P[mm * 32 + ln]      = (bf16)p0;             // C-layout -> LDS
      P[mm * 32 + 16 + ln] = (bf16)p1;
    }
    // Wave-local LDS round trip (C layout -> A layout); DS is in-order.
    asm volatile("s_wait_dscnt 0" ::: "memory");
    AFrag pa;
    pa.h[0] = *(const v8bf*)(P + ln * 32 + half * 16);
    pa.h[1] = *(const v8bf*)(P + ln * 32 + half * 16 + 8);
#pragma unroll
    for (int nt = 0; nt < 4; ++nt) {
      const bf16* vp = Vt + ((size_t)bh * HDc + nt * 16 + ln) * Tc + j0 + half * 16;
      o[nt] = wmma_bf16(pa.v, *(const v16bf*)vp, o[nt]);
    }
  }

  // Normalize and scatter to attn-out (bf16 [b*T][D]).
  const int bb = bh >> 4, h = bh & (Hc - 1);
#pragma unroll
  for (int r = 0; r < 8; ++r) {
    const int mm = r + half * 8;
    const float inv = 1.0f / lrow[r];
    const size_t ob = ((size_t)(bb * Tc + (qt0 + mm))) * Dc + h * HDc;
#pragma unroll
    for (int nt = 0; nt < 4; ++nt)
      AO[ob + nt * 16 + ln] = (bf16)(o[nt][r] * inv);
  }
}

// ---------------------------------------------------------------------------
extern "C" void kernel_launch(void* const* d_in, const int* in_sizes, int n_in,
                              void* d_out, int out_size, void* d_ws, size_t ws_size,
                              hipStream_t stream) {
  (void)in_sizes; (void)n_in; (void)out_size; (void)ws_size;
  const float* x     = (const float*)d_in[0];
  // d_in[1] = mask (implicit in causal kernel)
  const float* freqs = (const float*)d_in[2];
  const float* Wq = (const float*)d_in[3];  const float* bq = (const float*)d_in[4];
  const float* Wk = (const float*)d_in[5];  const float* bk = (const float*)d_in[6];
  const float* Wv = (const float*)d_in[7];  const float* bv = (const float*)d_in[8];
  const float* Wo = (const float*)d_in[9];  const float* bo = (const float*)d_in[10];

  char* ws = (char*)d_ws;                               // 40 MB used
  bf16* WqT = (bf16*)(ws + ((size_t)0  << 20));
  bf16* WkT = (bf16*)(ws + ((size_t)2  << 20));
  bf16* WvT = (bf16*)(ws + ((size_t)4  << 20));
  bf16* WoT = (bf16*)(ws + ((size_t)6  << 20));
  bf16* Qb  = (bf16*)(ws + ((size_t)8  << 20));
  bf16* Kb  = (bf16*)(ws + ((size_t)16 << 20));
  bf16* Vtb = (bf16*)(ws + ((size_t)24 << 20));
  bf16* Ab  = (bf16*)(ws + ((size_t)32 << 20));

  const int wblocks = (Dc * Dc) / 256;
  wcvt_kernel<<<wblocks, 256, 0, stream>>>(Wq, WqT);
  wcvt_kernel<<<wblocks, 256, 0, stream>>>(Wk, WkT);
  wcvt_kernel<<<wblocks, 256, 0, stream>>>(Wv, WvT);
  wcvt_kernel<<<wblocks, 256, 0, stream>>>(Wo, WoT);

  gemm16x64_kernel<0, 0><<<512, 256, 0, stream>>>((const void*)x, WqT, bq, (void*)Qb);
  gemm16x64_kernel<0, 0><<<512, 256, 0, stream>>>((const void*)x, WkT, bk, (void*)Kb);
  gemm16x64_kernel<0, 1><<<512, 256, 0, stream>>>((const void*)x, WvT, bv, (void*)Vtb);

  rope_kernel<<<(Bc * Hc * Tc * (HDc / 2)) / 256, 256, 0, stream>>>(Qb, Kb, freqs);

  attn_kernel<<<Bc * Hc * (Tc / 64), 128, 0, stream>>>(Qb, Kb, Vtb, Ab);

  gemm16x64_kernel<1, 2><<<512, 256, 0, stream>>>((const void*)Ab, WoT, bo, d_out);
}
